// BatchFlipLoss_85306640433706
// MI455X (gfx1250) — matrix operand order repaired and checked by reading, not compiled, not run on hardware
//
#include <hip/hip_runtime.h>

// ---------------------------------------------------------------------------
// BatchFlipLoss for MI455X (gfx1250, wave32, WMMA).
// N=4096, D=128, C=400, FLIP=8, T=0.1, ALPHA=0.03
// ---------------------------------------------------------------------------

#define FLIPC   8
#define NROWS   4096
#define DDIM    128
#define CCLS    400
#define BHALF   512          // rows per flip class
#define NPAIRS  36
#define INV_T   10.0f        // 1/T
#define ALPHA_W 0.03f

typedef __attribute__((ext_vector_type(16))) __bf16        v16bf;
typedef __attribute__((ext_vector_type(8)))  float         v8f;
typedef __attribute__((ext_vector_type(4)))  unsigned int  u32x4;

union Frag { v16bf v; u32x4 u[2]; };

// ---- ws layout (bytes) ----------------------------------------------------
// Fp   : bf16 [4096*128]              @ 0         (1,048,576 B)
// Rnd  : f32  [64*512]                @ 1,048,576 (  131,072 B)  row-sums excl diag
// Dg   : f32  [64*512]                @ 1,179,648 (  131,072 B)  block diagonals
// part : f32  [1152*4]                @ 1,310,720 (   18,432 B)  pass-B wave partials
// ceR  : f32  [4096]                  @ 1,329,152 (   16,384 B)  per-row CE loss
#define WS_RND   (1048576)
#define WS_DG    (1048576 + 131072)
#define WS_PART  (1310720)
#define WS_CER   (1329152)

__device__ inline unsigned short f32_to_bf16_rne(float f) {
  unsigned int u = __float_as_uint(f);
  unsigned int r = u + 0x7FFFu + ((u >> 16) & 1u);
  return (unsigned short)(r >> 16);
}

// A fragment: 16x32 bf16.  lane L: row m = L%16; lanes<16 hold K = ks*32+{0..7, 16..23},
// lanes>=16 hold K = ks*32+{8..15, 24..31}  (ISA 7.12.2, 16-bit A 16x32).
__device__ inline v16bf load_a_frag(const unsigned short* Fp, int rowBase, int lane, int ks) {
  int m     = lane & 15;
  int khalf = (lane < 16) ? 0 : 8;
  const unsigned short* p = Fp + (size_t)(rowBase + m) * DDIM + ks * 32;
  Frag f;
  f.u[0] = *(const u32x4*)(p + khalf);
  f.u[1] = *(const u32x4*)(p + 16 + khalf);
  return f.v;
}

// B fragment: 32x16 bf16 (column of Gram = row of Fp).  lane L: col n = L%16;
// lanes<16 hold K = ks*32+{0..7, 16..23}, lanes>=16 hold K = ks*32+{8..15, 24..31}.
__device__ inline v16bf load_b_frag(const unsigned short* Fp, int colBase, int lane, int ks) {
  int n    = lane & 15;
  int koff = (lane < 16) ? 0 : 8;
  const unsigned short* p = Fp + (size_t)(colBase + n) * DDIM + ks * 32 + koff;
  Frag f;
  f.u[0] = *(const u32x4*)(p);
  f.u[1] = *(const u32x4*)(p + 16);
  return f.v;
}

// ---------------------------------------------------------------------------
// K1: permute features into flip-class-major order and convert to bf16.
//     Fp[a*512 + k][d] = features[(k*8 + a)*128 + d]
// ---------------------------------------------------------------------------
__global__ void k_permute_bf16(const float* __restrict__ feat,
                               unsigned short* __restrict__ Fp) {
  int tid = blockIdx.x * blockDim.x + threadIdx.x;   // 524288 threads
  int q = tid >> 7;          // permuted row
  int d = tid & 127;
  int a = q >> 9;
  int k = q & 511;
  Fp[(size_t)q * DDIM + d] = f32_to_bf16_rne(feat[(size_t)(k * FLIPC + a) * DDIM + d]);
}

// ---------------------------------------------------------------------------
// K2: block diagonals Dg[a*8+b][k] = dot(features[k*8+a], features[k*8+b]) in f32.
// ---------------------------------------------------------------------------
__global__ void k_diag_dots(const float* __restrict__ feat, float* __restrict__ Dg) {
  int tid = blockIdx.x * blockDim.x + threadIdx.x;   // 32768 threads
  int ab = tid >> 9;
  int k  = tid & 511;
  int a = ab >> 3, b = ab & 7;
  const float4* pa = (const float4*)(feat + (size_t)(k * FLIPC + a) * DDIM);
  const float4* pb = (const float4*)(feat + (size_t)(k * FLIPC + b) * DDIM);
  float s = 0.f;
#pragma unroll 8
  for (int t = 0; t < DDIM / 4; ++t) {
    float4 x = pa[t], y = pb[t];
    s += x.x * y.x + x.y * y.y + x.z * y.z + x.w * y.w;
  }
  Dg[ab * BHALF + k] = s;
}

// ---------------------------------------------------------------------------
// K3 (pass A): per-block exp row-sums.
//   Rnd[ab][k] = sum_c exp(G_ab[k,c]/T), excluding c==k.
// Grid: 512 blocks x 128 thr (4 waves). Wave = one 16-row strip over 512 cols.
// ---------------------------------------------------------------------------
__global__ void k_block_rowsum(const unsigned short* __restrict__ Fp,
                               float* __restrict__ Rnd) {
  const int lane = threadIdx.x & 31;
  const int wave = threadIdx.x >> 5;
  const int ab   = blockIdx.x >> 3;          // 0..63
  const int wgi  = blockIdx.x & 7;
  const int a = ab >> 3, b = ab & 7;
  const int k0 = (wgi * 4 + wave) * 16;      // strip row base within block
  const int rowBase = a * BHALF + k0;
  const int colBase = b * BHALF;
  const int half8 = (lane < 16) ? 0 : 8;

  v16bf afr[4];
#pragma unroll
  for (int ks = 0; ks < 4; ++ks) afr[ks] = load_a_frag(Fp, rowBase, lane, ks);

  float rs[8];
#pragma unroll
  for (int v = 0; v < 8; ++v) rs[v] = 0.f;

  for (int n0 = 0; n0 < BHALF; n0 += 16) {
    v8f acc = {};
#pragma unroll
    for (int ks = 0; ks < 4; ++ks) {
      v16bf bfr = load_b_frag(Fp, colBase + n0, lane, ks);
      acc = __builtin_amdgcn_wmma_f32_16x16x32_bf16(
          false, afr[ks], false, bfr, (short)0, acc, false, false);
    }
    int cg = n0 + (lane & 15);               // column index within block
#pragma unroll
    for (int v = 0; v < 8; ++v) {
      int rk = k0 + v + half8;               // row index within block
      if (rk != cg) rs[v] += __expf(acc[v] * INV_T);
    }
  }
  // reduce across the 16 lanes sharing a row (columns)
#pragma unroll
  for (int v = 0; v < 8; ++v) {
#pragma unroll
    for (int m = 1; m <= 8; m <<= 1) rs[v] += __shfl_xor(rs[v], m, 32);
  }
  if (lane == 0) {
#pragma unroll
    for (int v = 0; v < 8; ++v) Rnd[ab * BHALF + k0 + v] = rs[v];
  } else if (lane == 16) {
#pragma unroll
    for (int v = 0; v < 8; ++v) Rnd[ab * BHALF + k0 + 8 + v] = rs[v];
  }
}

// ---------------------------------------------------------------------------
// K4 (pass B): per pair p=(i,j), 4 block positions, accumulate
//   sum log(1 - exp(G/T)/div_row), excluding the global diagonal (pos 0 and 3).
// Grid: 36*4*8 = 1152 blocks x 128 thr. Wave = 16-row strip; writes 1 partial.
// ---------------------------------------------------------------------------
__global__ void k_pair_logsum(const unsigned short* __restrict__ Fp,
                              const float* __restrict__ Rnd,
                              const float* __restrict__ Dg,
                              float* __restrict__ part) {
  const int lane = threadIdx.x & 31;
  const int wave = threadIdx.x >> 5;
  const int wgi  = blockIdx.x & 7;
  const int pos  = (blockIdx.x >> 3) & 3;
  const int p    = blockIdx.x >> 5;          // pair 0..35
  // decode (i,j), j>=i, python ordering
  int i = 0, rem = p;
  while (rem >= FLIPC - i) { rem -= (FLIPC - i); ++i; }
  const int j = i + rem;

  const int rowblk = (pos < 2) ? i : j;
  const int colblk = (pos == 0 || pos == 2) ? i : j;
  const bool excl  = (pos == 0) || (pos == 3);
  const bool first = (pos < 2);

  const int k0 = (wgi * 4 + wave) * 16;
  const int rowBase = rowblk * BHALF + k0;
  const int colBase = colblk * BHALF;
  const int half8 = (lane < 16) ? 0 : 8;

  // per-lane reciprocal denominators for its 8 rows
  float idv[8];
#pragma unroll
  for (int v = 0; v < 8; ++v) {
    int rk = k0 + v + half8;
    float dv;
    if (first) {
      dv = Rnd[(i * FLIPC + i) * BHALF + rk] + Rnd[(i * FLIPC + j) * BHALF + rk]
         + __expf(Dg[(i * FLIPC + j) * BHALF + rk] * INV_T);
    } else {
      dv = Rnd[(j * FLIPC + i) * BHALF + rk] + Rnd[(j * FLIPC + j) * BHALF + rk]
         + __expf(Dg[(j * FLIPC + i) * BHALF + rk] * INV_T);
    }
    idv[v] = 1.0f / dv;
  }

  v16bf afr[4];
#pragma unroll
  for (int ks = 0; ks < 4; ++ks) afr[ks] = load_a_frag(Fp, rowBase, lane, ks);

  float local = 0.f;
  for (int n0 = 0; n0 < BHALF; n0 += 16) {
    v8f acc = {};
#pragma unroll
    for (int ks = 0; ks < 4; ++ks) {
      v16bf bfr = load_b_frag(Fp, colBase + n0, lane, ks);
      acc = __builtin_amdgcn_wmma_f32_16x16x32_bf16(
          false, afr[ks], false, bfr, (short)0, acc, false, false);
    }
    int cg = n0 + (lane & 15);
#pragma unroll
    for (int v = 0; v < 8; ++v) {
      int rk = k0 + v + half8;
      if (!(excl && rk == cg))
        local += __logf(1.0f - __expf(acc[v] * INV_T) * idv[v]);
    }
  }
#pragma unroll
  for (int m = 1; m <= 16; m <<= 1) local += __shfl_xor(local, m, 32);
  if (lane == 0) part[blockIdx.x * 4 + wave] = local;
}

// ---------------------------------------------------------------------------
// K5: cross entropy per row (one wave per row). Grid 1024 x 128.
// ---------------------------------------------------------------------------
__global__ void k_cross_entropy(const float* __restrict__ predicts,
                                const int* __restrict__ labels,
                                float* __restrict__ ceR) {
  const int lane = threadIdx.x & 31;
  const int wave = threadIdx.x >> 5;
  const int row  = blockIdx.x * 4 + wave;
  const float* pr = predicts + (size_t)row * CCLS;

  float mx = -1e30f;
  for (int c = lane; c < CCLS; c += 32) mx = fmaxf(mx, pr[c]);
#pragma unroll
  for (int m = 1; m <= 16; m <<= 1) mx = fmaxf(mx, __shfl_xor(mx, m, 32));

  float se = 0.f;
  for (int c = lane; c < CCLS; c += 32) se += __expf(pr[c] - mx);
#pragma unroll
  for (int m = 1; m <= 16; m <<= 1) se += __shfl_xor(se, m, 32);

  if (lane == 0) ceR[row] = __logf(se) + mx - pr[labels[row]];
}

// ---------------------------------------------------------------------------
// K6: final assembly. Single block of 256 threads:
//  - sum pass-B partials (4608)
//  - per-pair per-row terms: log(pos) - log(div) - log(1 - pos/div), both halves
//  - sum CE rows, combine: out = ALPHA * (-S/1024) + CE/4096
// ---------------------------------------------------------------------------
__global__ void k_finalize(const float* __restrict__ Rnd,
                           const float* __restrict__ Dg,
                           const float* __restrict__ part,
                           const float* __restrict__ ceR,
                           float* __restrict__ out) {
  __shared__ float sA[256];
  __shared__ float sB[256];
  const int t = threadIdx.x;

  float s = 0.f;
  for (int idx = t; idx < 1152 * 4; idx += 256) s += part[idx];

  for (int q = t; q < NPAIRS * BHALF; q += 256) {
    int p = q >> 9, k = q & 511;
    int i = 0, rem = p;
    while (rem >= FLIPC - i) { rem -= (FLIPC - i); ++i; }
    int j = i + rem;
    float dij = Dg[(i * FLIPC + j) * BHALF + k];
    float lp  = dij * INV_T;                 // log(pos)
    float pe  = __expf(lp);
    float d1 = Rnd[(i * FLIPC + i) * BHALF + k] + Rnd[(i * FLIPC + j) * BHALF + k] + pe;
    float d2 = Rnd[(j * FLIPC + i) * BHALF + k] + Rnd[(j * FLIPC + j) * BHALF + k] + pe;
    s += lp - __logf(d1) - __logf(1.0f - pe / d1);
    s += lp - __logf(d2) - __logf(1.0f - pe / d2);
  }

  float ce = 0.f;
  for (int idx = t; idx < NROWS; idx += 256) ce += ceR[idx];

  sA[t] = s; sB[t] = ce;
  __syncthreads();
  for (int w = 128; w > 0; w >>= 1) {
    if (t < w) { sA[t] += sA[t + w]; sB[t] += sB[t + w]; }
    __syncthreads();
  }
  if (t == 0)
    out[0] = ALPHA_W * (-(sA[0]) / 1024.0f) + sB[0] / (float)NROWS;
}

// ---------------------------------------------------------------------------
extern "C" void kernel_launch(void* const* d_in, const int* in_sizes, int n_in,
                              void* d_out, int out_size, void* d_ws, size_t ws_size,
                              hipStream_t stream) {
  const float* predicts = (const float*)d_in[0];   // [4096,400]
  const int*   labels   = (const int*)d_in[1];     // [4096]
  const float* features = (const float*)d_in[2];   // [4096,128]
  (void)d_in; (void)in_sizes; (void)n_in; (void)out_size; (void)ws_size;

  unsigned short* Fp  = (unsigned short*)d_ws;
  float* Rnd  = (float*)((char*)d_ws + WS_RND);
  float* Dg   = (float*)((char*)d_ws + WS_DG);
  float* part = (float*)((char*)d_ws + WS_PART);
  float* ceR  = (float*)((char*)d_ws + WS_CER);
  float* out  = (float*)d_out;

  k_permute_bf16 <<<2048, 256, 0, stream>>>(features, Fp);
  k_diag_dots    <<<128,  256, 0, stream>>>(features, Dg);
  k_block_rowsum <<<512,  128, 0, stream>>>(Fp, Rnd);
  k_pair_logsum  <<<1152, 128, 0, stream>>>(Fp, Rnd, Dg, part);
  k_cross_entropy<<<1024, 128, 0, stream>>>(predicts, labels, ceR);
  k_finalize     <<<1,    256, 0, stream>>>(Rnd, Dg, part, ceR, out);
}
